// FusedSelfAttention_13477607374956
// MI455X (gfx1250) — matrix operand order repaired and checked by reading, not compiled, not run on hardware
//
#include <hip/hip_runtime.h>
#include <cstdint>

// ---------------- problem constants ----------------
#define BB   2
#define TT   2048
#define DD   1024
#define HH   16
#define DH   64
#define MM   (BB * TT)          // 4096 rows
#define N3   (3 * DD)           // 3072

#define BROW 40                 // LDS B-tile row stride in bf16 elems (80B, pad kills bank conflicts)

typedef __attribute__((ext_vector_type(16))) __bf16 v16bf;
typedef __attribute__((ext_vector_type(8)))  __bf16 v8bf;
typedef __attribute__((ext_vector_type(8)))  float  v8f;

// float -> bf16 (round to nearest even)
__device__ __forceinline__ uint16_t f2bf(float f) {
    union { float f; uint32_t u; } v; v.f = f;
    uint32_t r = v.u + 0x7FFFu + ((v.u >> 16) & 1u);
    return (uint16_t)(r >> 16);
}

// A-fragment: 16x32 bf16 tile from row-major [ld] matrix (contiguous k order: k = 16*half + s).
__device__ __forceinline__ v16bf load_a_frag(const uint16_t* p, int ld, int row, int col) {
    int lane = threadIdx.x & 31;
    const uint16_t* base = p + (size_t)(row + (lane & 15)) * ld + col + ((lane >> 4) << 4);
    v8bf lo = *(const v8bf*)(base);
    v8bf hi = *(const v8bf*)(base + 8);
    return __builtin_shufflevector(lo, hi, 0,1,2,3,4,5,6,7,8,9,10,11,12,13,14,15);
}

// B-fragment (K=32 x N=16) from the TRANSPOSE of B in global memory: bt is [n][k], stride ld.
__device__ __forceinline__ v16bf load_bt_frag(const uint16_t* bt, int ld, int n0, int k0) {
    int lane = threadIdx.x & 31;
    const uint16_t* base = bt + (size_t)(n0 + (lane & 15)) * ld + k0 + ((lane >> 4) << 4);
    v8bf lo = *(const v8bf*)(base);
    v8bf hi = *(const v8bf*)(base + 8);
    return __builtin_shufflevector(lo, hi, 0,1,2,3,4,5,6,7,8,9,10,11,12,13,14,15);
}

// B-fragment read from an LDS-staged 64xBROW tile (n-major), sub-tile t (n = 16t..16t+15)
__device__ __forceinline__ v16bf load_b_frag_lds(const uint16_t* buf, int t) {
    int lane = threadIdx.x & 31;
    const uint16_t* base = buf + (size_t)(t * 16 + (lane & 15)) * BROW + ((lane >> 4) << 4);
    v8bf lo = *(const v8bf*)(base);
    v8bf hi = *(const v8bf*)(base + 8);
    return __builtin_shufflevector(lo, hi, 0,1,2,3,4,5,6,7,8,9,10,11,12,13,14,15);
}

__device__ __forceinline__ v8f wmma_bf16(v16bf a, v16bf b, v8f c) {
    return __builtin_amdgcn_wmma_f32_16x16x32_bf16(false, a, false, b, (short)0, c, false, false);
}

// Cooperative async stage of a 64(n) x 32(k) bf16 B-slice into LDS (4KB payload).
// 128 lanes x 32B each, via gfx1250 async global->LDS DMA (tracked by ASYNCcnt).
__device__ __forceinline__ void stage_b_async(const uint16_t* bt, int ld, int n0, int k0,
                                              uint16_t* buf) {
    int tid  = threadIdx.x;
    int row  = tid >> 1;          // 0..63
    int half = tid & 1;           // two 32B halves of a 64B row
    const uint16_t* g = bt + (size_t)(n0 + row) * ld + k0 + half * 16;
    uint32_t lo = (uint32_t)(uintptr_t)(buf + row * BROW + half * 16);
    asm volatile("global_load_async_to_lds_b128 %0, %1, off"
                 :: "v"(lo), "v"(g) : "memory");
    asm volatile("global_load_async_to_lds_b128 %0, %1, off offset:16"
                 :: "v"(lo), "v"(g) : "memory");
}

__device__ __forceinline__ void wait_async_all() {
    asm volatile("s_wait_asynccnt 0x0" ::: "memory");
}

// ---------------- kernel 1: convert / transpose to bf16 ----------------
__global__ void cvt_kernel(const float* __restrict__ x, const float* __restrict__ wqkv,
                           const float* __restrict__ wo,
                           uint16_t* __restrict__ xbf, uint16_t* __restrict__ wqkvT,
                           uint16_t* __restrict__ woT) {
    const int64_t XN  = (int64_t)MM * DD;
    const int64_t WQN = (int64_t)DD * N3;
    const int64_t WON = (int64_t)DD * DD;
    int64_t stride = (int64_t)gridDim.x * blockDim.x;
    int64_t tid = (int64_t)blockIdx.x * blockDim.x + threadIdx.x;
    for (int64_t i = tid; i < XN; i += stride) xbf[i] = f2bf(x[i]);
    for (int64_t i = tid; i < WQN; i += stride) {
        int c = (int)(i >> 10), k = (int)(i & 1023);      // wqkvT[c][k] = wqkv[k][c]
        wqkvT[i] = f2bf(wqkv[(int64_t)k * N3 + c]);
    }
    for (int64_t i = tid; i < WON; i += stride) {
        int n = (int)(i >> 10), k = (int)(i & 1023);      // woT[n][k] = wo[k][n]
        woT[i] = f2bf(wo[((int64_t)k << 10) + n]);
    }
}

// ---------------- kernel 2: QKV GEMM, LDS-staged B, 32x64 tile per wave ----------------
// block = 128 threads (4 waves) -> 128(M) x 64(N) output; B slice shared via async LDS stage.
__global__ __launch_bounds__(128) void qkv_gemm_kernel(const uint16_t* __restrict__ xbf,
                                                       const uint16_t* __restrict__ wqkvT,
                                                       uint16_t* __restrict__ Qb,
                                                       uint16_t* __restrict__ Kb,
                                                       uint16_t* __restrict__ Vt) {
    __shared__ __align__(16) uint16_t bsm[2 * 64 * BROW];
    int wave = threadIdx.x >> 5, lane = threadIdx.x & 31;
    const int NT = N3 / 64;                       // 48 N-tiles
    int bm = blockIdx.x / NT, bn = blockIdx.x % NT;
    int mrow = bm * 128 + wave * 32;
    int n0 = bn * 64;

    v8f acc0[4] = {v8f{}, v8f{}, v8f{}, v8f{}};
    v8f acc1[4] = {v8f{}, v8f{}, v8f{}, v8f{}};

    stage_b_async(wqkvT, DD, n0, 0, bsm);
    v16bf a0 = load_a_frag(xbf, DD, mrow,      0);
    v16bf a1 = load_a_frag(xbf, DD, mrow + 16, 0);
    wait_async_all();
    __syncthreads();

    for (int k0 = 0; k0 < DD; k0 += 32) {
        uint16_t* cur = bsm + ((k0 >> 5) & 1) * (64 * BROW);
        uint16_t* nxt = bsm + (((k0 >> 5) + 1) & 1) * (64 * BROW);
        bool more = (k0 + 32) < DD;
        if (more) stage_b_async(wqkvT, DD, n0, k0 + 32, nxt);
        v16bf na0 = a0, na1 = a1;
        if (more) {
            na0 = load_a_frag(xbf, DD, mrow,      k0 + 32);
            na1 = load_a_frag(xbf, DD, mrow + 16, k0 + 32);
        }
        #pragma unroll
        for (int t = 0; t < 4; ++t) {
            v16bf bf = load_b_frag_lds(cur, t);
            acc0[t] = wmma_bf16(a0, bf, acc0[t]);
            acc1[t] = wmma_bf16(a1, bf, acc1[t]);
        }
        a0 = na0; a1 = na1;
        if (more) wait_async_all();
        __syncthreads();
    }

    int halfl = lane >> 4, cl = lane & 15;
    #pragma unroll
    for (int mi = 0; mi < 2; ++mi) {
        #pragma unroll
        for (int t = 0; t < 4; ++t) {
            #pragma unroll
            for (int v = 0; v < 8; ++v) {
                int m  = mrow + mi * 16 + v + halfl * 8;
                int b  = m >> 11, tq = m & (TT - 1);
                int c  = n0 + t * 16 + cl;
                int h  = c / 192, w = c % 192;
                int kind = w >> 6, dhi = w & 63;
                size_t bh = (size_t)b * HH + h;
                float val = (mi == 0) ? acc0[t][v] : acc1[t][v];
                if (kind == 0)       Qb[(bh * TT + tq) * DH + dhi] = f2bf(val * 0.03125f);
                else if (kind == 1)  Kb[(bh * TT + tq) * DH + dhi] = f2bf(val);
                else                 Vt[(bh * DH + dhi) * TT + tq] = f2bf(val);
            }
        }
    }
}

// ---------------- kernel 3: flash attention (one wave = one (b,h,16-query tile)) ----------
__global__ __launch_bounds__(128) void attn_kernel(const uint16_t* __restrict__ Qb,
                                                   const uint16_t* __restrict__ Kb,
                                                   const uint16_t* __restrict__ Vt,
                                                   uint16_t* __restrict__ attn) {
    __shared__ __align__(16) uint16_t smem[4 * 16 * 32];
    int wave = threadIdx.x >> 5, lane = threadIdx.x & 31;
    int task = blockIdx.x * 4 + wave;            // BB*HH*(TT/16) = 4096 tasks
    int qt = task & (TT / 16 - 1);
    int bh = task >> 7;
    int q0 = qt * 16;
    const uint16_t* Qp = Qb + (size_t)bh * TT * DH;
    const uint16_t* Kp = Kb + (size_t)bh * TT * DH;
    const uint16_t* Vp = Vt + (size_t)bh * DH * TT;
    uint16_t* pl = smem + wave * 16 * 32;

    v16bf aq0 = load_a_frag(Qp, DH, q0, 0);
    v16bf aq1 = load_a_frag(Qp, DH, q0, 32);

    int half = lane >> 4, cl = lane & 15;
    float mrow[8], lrow[8];
    v8f o[4] = {v8f{}, v8f{}, v8f{}, v8f{}};
    #pragma unroll
    for (int v = 0; v < 8; ++v) { mrow[v] = -1e30f; lrow[v] = 0.f; }

    for (int j0 = 0; j0 <= q0 + 15; j0 += 32) {
        v8f s0{}, s1{};
        s0 = wmma_bf16(aq0, load_bt_frag(Kp, DH, j0,      0),  s0);
        s0 = wmma_bf16(aq1, load_bt_frag(Kp, DH, j0,      32), s0);
        s1 = wmma_bf16(aq0, load_bt_frag(Kp, DH, j0 + 16, 0),  s1);
        s1 = wmma_bf16(aq1, load_bt_frag(Kp, DH, j0 + 16, 32), s1);

        if (j0 + 31 > q0) {                       // causal boundary block
            #pragma unroll
            for (int v = 0; v < 8; ++v) {
                int q = q0 + v + half * 8;
                if (j0 + cl > q)      s0[v] = -1e30f;
                if (j0 + 16 + cl > q) s1[v] = -1e30f;
            }
        }

        #pragma unroll
        for (int v = 0; v < 8; ++v) {
            float t = fmaxf(s0[v], s1[v]);
            t = fmaxf(t, __shfl_xor(t, 1, 32));
            t = fmaxf(t, __shfl_xor(t, 2, 32));
            t = fmaxf(t, __shfl_xor(t, 4, 32));
            t = fmaxf(t, __shfl_xor(t, 8, 32));
            float mn = fmaxf(mrow[v], t);
            float al = __expf(mrow[v] - mn);
            float p0 = __expf(s0[v] - mn);
            float p1 = __expf(s1[v] - mn);
            s0[v] = p0; s1[v] = p1;
            float rs = p0 + p1;
            rs += __shfl_xor(rs, 1, 32);
            rs += __shfl_xor(rs, 2, 32);
            rs += __shfl_xor(rs, 4, 32);
            rs += __shfl_xor(rs, 8, 32);
            lrow[v] = lrow[v] * al + rs;
            mrow[v] = mn;
            o[0][v] *= al; o[1][v] *= al; o[2][v] *= al; o[3][v] *= al;
        }

        // stage P (bf16, 16x32 row-major) in LDS: C-layout -> A-fragment reshape
        #pragma unroll
        for (int v = 0; v < 8; ++v) {
            int r = v + half * 8;
            pl[r * 32 + cl]      = f2bf(s0[v]);
            pl[r * 32 + 16 + cl] = f2bf(s1[v]);
        }
        asm volatile("s_wait_dscnt 0x0" ::: "memory");
        v16bf ap = load_a_frag(pl, 32, 0, 0);

        #pragma unroll
        for (int t = 0; t < 4; ++t)
            o[t] = wmma_bf16(ap, load_bt_frag(Vp, TT, t * 16, j0), o[t]);
    }

    int b = bh >> 4, h = bh & 15;
    #pragma unroll
    for (int v = 0; v < 8; ++v) {
        float inv = 1.f / lrow[v];
        size_t row = (size_t)b * TT + q0 + v + half * 8;
        #pragma unroll
        for (int t = 0; t < 4; ++t)
            attn[row * DD + h * DH + t * 16 + cl] = f2bf(o[t][v] * inv);
    }
}

// ---------------- kernel 4: output GEMM, same LDS-staged structure -> fp32 ----------------
__global__ __launch_bounds__(128) void out_gemm_kernel(const uint16_t* __restrict__ attn,
                                                       const uint16_t* __restrict__ woT,
                                                       float* __restrict__ out) {
    __shared__ __align__(16) uint16_t bsm[2 * 64 * BROW];
    int wave = threadIdx.x >> 5, lane = threadIdx.x & 31;
    const int NT = DD / 64;                       // 16 N-tiles
    int bm = blockIdx.x / NT, bn = blockIdx.x % NT;
    int mrow = bm * 128 + wave * 32;
    int n0 = bn * 64;

    v8f acc0[4] = {v8f{}, v8f{}, v8f{}, v8f{}};
    v8f acc1[4] = {v8f{}, v8f{}, v8f{}, v8f{}};

    stage_b_async(woT, DD, n0, 0, bsm);
    v16bf a0 = load_a_frag(attn, DD, mrow,      0);
    v16bf a1 = load_a_frag(attn, DD, mrow + 16, 0);
    wait_async_all();
    __syncthreads();

    for (int k0 = 0; k0 < DD; k0 += 32) {
        uint16_t* cur = bsm + ((k0 >> 5) & 1) * (64 * BROW);
        uint16_t* nxt = bsm + (((k0 >> 5) + 1) & 1) * (64 * BROW);
        bool more = (k0 + 32) < DD;
        if (more) stage_b_async(woT, DD, n0, k0 + 32, nxt);
        v16bf na0 = a0, na1 = a1;
        if (more) {
            na0 = load_a_frag(attn, DD, mrow,      k0 + 32);
            na1 = load_a_frag(attn, DD, mrow + 16, k0 + 32);
        }
        #pragma unroll
        for (int t = 0; t < 4; ++t) {
            v16bf bf = load_b_frag_lds(cur, t);
            acc0[t] = wmma_bf16(a0, bf, acc0[t]);
            acc1[t] = wmma_bf16(a1, bf, acc1[t]);
        }
        a0 = na0; a1 = na1;
        if (more) wait_async_all();
        __syncthreads();
    }

    int halfl = lane >> 4, cl = lane & 15;
    #pragma unroll
    for (int mi = 0; mi < 2; ++mi)
        #pragma unroll
        for (int t = 0; t < 4; ++t)
            #pragma unroll
            for (int v = 0; v < 8; ++v)
                out[(size_t)(mrow + mi * 16 + v + halfl * 8) * DD + n0 + t * 16 + cl] =
                    (mi == 0) ? acc0[t][v] : acc1[t][v];
}

// ---------------- launch ----------------
extern "C" void kernel_launch(void* const* d_in, const int* in_sizes, int n_in,
                              void* d_out, int out_size, void* d_ws, size_t ws_size,
                              hipStream_t stream) {
    const float* x    = (const float*)d_in[0];
    const float* wqkv = (const float*)d_in[1];
    const float* wo   = (const float*)d_in[2];
    float* out = (float*)d_out;

    char* w = (char*)d_ws;
    uint16_t* xbf   = (uint16_t*)w;  w += (size_t)MM * DD * 2;
    uint16_t* wqkvT = (uint16_t*)w;  w += (size_t)DD * N3 * 2;
    uint16_t* woT   = (uint16_t*)w;  w += (size_t)DD * DD * 2;
    uint16_t* Qb    = (uint16_t*)w;  w += (size_t)MM * DD * 2;
    uint16_t* Kb    = (uint16_t*)w;  w += (size_t)MM * DD * 2;
    uint16_t* Vt    = (uint16_t*)w;  w += (size_t)MM * DD * 2;
    uint16_t* attn  = (uint16_t*)w;  w += (size_t)MM * DD * 2;

    cvt_kernel<<<2048, 256, 0, stream>>>(x, wqkv, wo, xbf, wqkvT, woT);

    // (MM/128)=32 M-blocks x (N3/64)=48 N-blocks
    qkv_gemm_kernel<<<32 * 48, 128, 0, stream>>>(xbf, wqkvT, Qb, Kb, Vt);

    // 2*16*128 = 4096 wave-tasks / 4
    attn_kernel<<<1024, 128, 0, stream>>>(Qb, Kb, Vt, attn);

    // (MM/128)=32 x (DD/64)=16
    out_gemm_kernel<<<32 * 16, 128, 0, stream>>>(attn, woT, out);
}
